// FourierLayer_7267084665111
// MI455X (gfx1250) — matrix operand order repaired and checked by reading, not compiled, not run on hardware
//
#include <hip/hip_runtime.h>
#include <hip/hip_bf16.h>
#include <math.h>

// FourierLayer for MI455X (gfx1250): all heavy math as WMMA GEMMs.
//   B=8, CIN=COUT=128, H=W=256, modes 64x33 (padded to 64x48).
// Pipeline: tables -> pack W(bf16) -> W-rDFT (TDM-staged) -> H-DFT
//           -> spectral mix (bf16 WMMA) -> H-iDFT
//           -> fused (skip GEMM + W-irDFT + bias + GELU).
// Workspace use: ~277 MB of d_ws.

#define BATCH 8
#define CIN   128
#define COUT  128
#define HH    256
#define WW    256
#define M1    64
#define M2    33
#define NP    48            // spectral width padded to 3 N-tiles
#define NMODE (M1*M2)       // 2112

typedef __attribute__((ext_vector_type(2)))  float  v2f;
typedef __attribute__((ext_vector_type(8)))  float  v8f;
typedef __attribute__((ext_vector_type(16))) __bf16 v16bf;
typedef __attribute__((ext_vector_type(4)))  unsigned int v4u;
typedef __attribute__((ext_vector_type(4)))  int    v4i;
typedef __attribute__((ext_vector_type(8)))  int    v8i;

__device__ __forceinline__ v8f wmma_f32(v2f a, v2f b, v8f c) {
  // V_WMMA_F32_16X16X4_F32 : 16x4 A (2 VGPR), 4x16 B (2 VGPR), f32 C/D (8 VGPR)
  return __builtin_amdgcn_wmma_f32_16x16x4_f32(false, a, false, b, (short)0, c, false, false);
}
__device__ __forceinline__ v8f wmma_bf16(v16bf a, v16bf b, v8f c) {
  // V_WMMA_F32_16X16X32_BF16
  return __builtin_amdgcn_wmma_f32_16x16x32_bf16(false, a, false, b, (short)0, c, false, false);
}

__device__ __forceinline__ float gelu_exact(float v) {
  return 0.5f * v * (1.0f + erff(v * 0.70710678118654752f));
}

// ---------------------------------------------------------------- stage 0: DFT tables
__global__ void k_tables(float* F1, float* Fhr, float* Fhi,
                         float* Ghr, float* Ghi, float* G2) {
  const float PI2 = 6.283185307179586f;
  int t = threadIdx.x;
  // F1[k][j], j<48: re col n=j ; j>=48: im col n=j-48.  Forward norm 1/(H*W) baked in.
  for (int idx = t; idx < 256 * 96; idx += 256) {
    int k = idx / 96, j = idx % 96, n = (j < 48) ? j : j - 48;
    float v = 0.f;
    if (n < M2) {
      float th = PI2 * (float)(k * n) / 256.f;
      v = (j < 48) ? cosf(th) : -sinf(th);
      v *= (1.0f / 65536.0f);
    }
    F1[idx] = v;
  }
  // Fh[m][h] = exp(-2*pi*i*(m-32)*h/256)   (fftshifted row slice -> freqs -32..31)
  for (int idx = t; idx < M1 * 256; idx += 256) {
    int m = idx / 256, h = idx % 256;
    float th = PI2 * (float)((m - 32) * h) / 256.f;
    Fhr[idx] = cosf(th); Fhi[idx] = -sinf(th);
  }
  // Gh[h][m] = exp(+2*pi*i*(m-32)*h/256)
  for (int idx = t; idx < 256 * M1; idx += 256) {
    int h = idx / M1, m = idx % M1;
    float th = PI2 * (float)((m - 32) * h) / 256.f;
    Ghr[idx] = cosf(th); Ghi[idx] = sinf(th);
  }
  // G2[j][w]: inverse rDFT with hermitian weights c_n (c_0=1, else 2); j<48 re, j>=48 im.
  for (int idx = t; idx < 96 * 256; idx += 256) {
    int j = idx / 256, w = idx % 256, n = (j < 48) ? j : j - 48;
    float v = 0.f;
    if (n < M2) {
      float c = (n == 0) ? 1.f : 2.f;
      float th = PI2 * (float)(n * w) / 256.f;
      v = (j < 48) ? c * cosf(th) : -c * sinf(th);
    }
    G2[idx] = v;
  }
}

// ------------------------------------------- stage 0b: pack weights -> bf16 [mode][k=256][o]
// k<128 -> w_r[i=k], k>=128 -> w_i[i=k-128]. LDS tile transpose: coalesced in and out.
__global__ void __launch_bounds__(256) k_pack(const float* __restrict__ wr,
                                              const float* __restrict__ wi,
                                              __bf16* __restrict__ Wp) {
  int i2 = blockIdx.x;  // k index 0..255
  int mb = blockIdx.y;  // mode block 0..32 (64 modes each)
  const float* src = (i2 < CIN) ? (wr + (size_t)i2 * COUT * NMODE)
                                : (wi + (size_t)(i2 - CIN) * COUT * NMODE);
  __shared__ float tile[128 * 64];
  int t = threadIdx.x;
  for (int e = t; e < 128 * 64; e += 256) {
    int o = e >> 6, md = e & 63;
    tile[o * 64 + md] = src[(size_t)o * NMODE + mb * 64 + md];
  }
  __syncthreads();
  for (int e = t; e < 128 * 64; e += 256) {
    int md = e >> 7, o = e & 127;
    int mode = mb * 64 + md;
    Wp[((size_t)mode * 256 + i2) * COUT + o] = (__bf16)tile[o * 64 + md];
  }
}

// ---------------------------------------------------- stage 1: W-direction partial rDFT
// rows (b,i,h) x [256] @ F1[256x96] -> Xw re/im planes [row][48]
// x tile (32x256 f32) is staged to LDS by the Tensor Data Mover with a 4-DWORD
// row pad (256 DWORD interval), reproducing the xt[32][260] anti-bank-conflict
// layout; TENSORcnt + workgroup barrier order it against the WMMA consumers.
__global__ void __launch_bounds__(64) k_stage1(const float* __restrict__ x,
                                               const float* __restrict__ F1,
                                               float* __restrict__ Xwr,
                                               float* __restrict__ Xwi) {
  __shared__ float xt[32 * 260];   // 32 rows, padded stride 260 to break bank conflicts
  int t = threadIdx.x;
  size_t row0 = (size_t)blockIdx.x * 32;

#if __has_builtin(__builtin_amdgcn_tensor_load_to_lds) && \
    __has_builtin(__builtin_amdgcn_s_wait_tensorcnt)
  if (t < 32) {                                   // wave 0 drives the TDM (EXEC ignored)
    unsigned long long ga = (unsigned long long)(uintptr_t)(x + row0 * 256);
    unsigned int lds_off = (unsigned int)(uintptr_t)(&xt[0]);   // low 32 bits = LDS addr
    v4u g0;                                       // D# group 0
    g0[0] = 1u;                                   // count=1 (valid), user mode
    g0[1] = lds_off;                              // lds_addr (bytes)
    g0[2] = (unsigned int)(ga & 0xffffffffu);     // global_addr[31:0]
    g0[3] = (unsigned int)((ga >> 32) & 0x1ffffffu) | (2u << 30);  // addr[56:32] | type=2
    v8i g1;                                       // D# group 1
    // data_size=4B(2), pad_enable, pad_interval=256 DWORDs(7), pad_amount=4 DWORDs(3)
    g1[0] = (int)((2u << 16) | (1u << 20) | (7u << 22) | (3u << 25));
    g1[1] = (int)(256u << 16);                    // tensor_dim0 = 256 (bits 79:48)
    g1[2] = (int)(32u << 16);                     // tensor_dim1 = 32  (bits 111:80)
    g1[3] = (int)(256u << 16);                    // tile_dim0   = 256 (bits 127:112)
    g1[4] = 32;                                   // tile_dim1 = 32 (tile_dim2 = 0)
    g1[5] = 256;                                  // tensor_dim0_stride = 256
    g1[6] = 0;                                    // dim0_stride hi / dim1_stride lo
    g1[7] = 0;                                    // tensor_dim1_stride hi
    v4i g2 = {0, 0, 0, 0};                        // 2D tensor: groups 2/3 unused
    v4i g3 = {0, 0, 0, 0};
#if defined(__clang_major__) && (__clang_major__ >= 23)
    v8i g4 = {0, 0, 0, 0, 0, 0, 0, 0};
    __builtin_amdgcn_tensor_load_to_lds(g0, g1, g2, g3, g4, 0);
#else
    __builtin_amdgcn_tensor_load_to_lds(g0, g1, g2, g3, 0);
#endif
    __builtin_amdgcn_s_wait_tensorcnt(0);         // s_wait_tensorcnt 0
  }
  __syncthreads();
#else
  {
    const float4* xs4 = (const float4*)(x + row0 * 256);
    for (int e = t; e < 32 * 64; e += 64) {
      int r = e >> 6, c = e & 63;
      float4 v = xs4[r * 64 + c];
      float* d = &xt[r * 260 + c * 4];
      d[0] = v.x; d[1] = v.y; d[2] = v.z; d[3] = v.w;
    }
    __syncthreads();
  }
#endif

  int lane = t & 31, wave = t >> 5;
  int m = lane & 15, hi = lane >> 4, kb = hi * 2, n = lane & 15;
  for (int nt = 0; nt < 6; ++nt) {
    v8f acc = {};
    for (int k0 = 0; k0 < 256; k0 += 4) {
      v2f a, b;
      a.x = xt[(wave * 16 + m) * 260 + k0 + kb];
      a.y = xt[(wave * 16 + m) * 260 + k0 + kb + 1];
      b.x = F1[(k0 + kb) * 96 + nt * 16 + n];
      b.y = F1[(k0 + kb + 1) * 96 + nt * 16 + n];
      acc = wmma_f32(a, b, acc);
    }
    for (int r = 0; r < 8; ++r) {
      size_t grow = row0 + wave * 16 + r + hi * 8;
      int j = nt * 16 + n;
      if (j < NP) Xwr[grow * NP + j] = acc[r];
      else        Xwi[grow * NP + (j - NP)] = acc[r];
    }
  }
}

// ------------------------------------------------- stage 2: H-direction partial DFT (complex)
// per (b,i): Xs = Fh @ Xw  (64x256x48, 4 real accumulations)
__global__ void __launch_bounds__(128) k_stage2(const float* __restrict__ Xwr,
                                                const float* __restrict__ Xwi,
                                                const float* __restrict__ Fhr,
                                                const float* __restrict__ Fhi,
                                                float* __restrict__ Xsr,
                                                float* __restrict__ Xsi) {
  int bi = blockIdx.x;
  const float* brb = Xwr + (size_t)bi * 256 * NP;
  const float* bib = Xwi + (size_t)bi * 256 * NP;
  int t = threadIdx.x, lane = t & 31, wave = t >> 5;
  int m = lane & 15, hi = lane >> 4, kb = hi * 2, n = lane & 15;
  int m0 = wave * 16;
  for (int nt = 0; nt < 3; ++nt) {
    v8f arr = {}, aii = {}, ari = {}, air = {};
    for (int k0 = 0; k0 < 256; k0 += 4) {
      v2f ar, ai, br, bi2;
      ar.x = Fhr[(m0 + m) * 256 + k0 + kb]; ar.y = Fhr[(m0 + m) * 256 + k0 + kb + 1];
      ai.x = Fhi[(m0 + m) * 256 + k0 + kb]; ai.y = Fhi[(m0 + m) * 256 + k0 + kb + 1];
      br.x = brb[(k0 + kb) * NP + nt * 16 + n];  br.y = brb[(k0 + kb + 1) * NP + nt * 16 + n];
      bi2.x = bib[(k0 + kb) * NP + nt * 16 + n]; bi2.y = bib[(k0 + kb + 1) * NP + nt * 16 + n];
      arr = wmma_f32(ar, br, arr);  aii = wmma_f32(ai, bi2, aii);
      ari = wmma_f32(ar, bi2, ari); air = wmma_f32(ai, br, air);
    }
    for (int r = 0; r < 8; ++r) {
      int mc = m0 + r + hi * 8, nc = nt * 16 + n;
      size_t addr = ((size_t)bi * M1 + mc) * NP + nc;
      Xsr[addr] = arr[r] - aii[r];
      Xsi[addr] = ari[r] + air[r];
    }
  }
}

// -------------------------------------------- stage 3: per-mode spectral mix (bf16 WMMA)
// A = [[Xr,-Xi],[Xi,Xr]] (16x256), B = [Wr;Wi] packed bf16 -> C top=re, bottom=im
__global__ void __launch_bounds__(256) k_stage3(const float* __restrict__ Xsr,
                                                const float* __restrict__ Xsi,
                                                const __bf16* __restrict__ Wp,
                                                float* __restrict__ Ysr,
                                                float* __restrict__ Ysi) {
  int mode = blockIdx.x;
  int mm = mode / M2, nn = mode - mm * M2;
  __shared__ __bf16 As[16 * 256];
  int t = threadIdx.x;
  if (t < CIN) {
    int i = t;
    for (int b = 0; b < BATCH; ++b) {
      size_t a = (((size_t)(b * CIN + i)) * M1 + mm) * NP + nn;
      float xr = Xsr[a], xi = Xsi[a];
      As[b * 256 + i]             = (__bf16)xr;
      As[b * 256 + 128 + i]       = (__bf16)(-xi);
      As[(b + 8) * 256 + i]       = (__bf16)xi;
      As[(b + 8) * 256 + 128 + i] = (__bf16)xr;
    }
  }
  __syncthreads();
  int lane = t & 31, wave = t >> 5, o0 = wave * 16;
  int mrow = lane & 15, hiA = (lane >> 4) * 8;
  int ncol = lane & 15, hiB = (lane >> 4) * 8;
  v8f acc = {};
  for (int k0 = 0; k0 < 256; k0 += 32) {
    v16bf a, b;
    for (int r = 0; r < 8; ++r) {
      int ka = k0 + 2 * r + ((r >= 4) ? 8 : 0) + hiA;   // 16-bit A layout (ISA 7.12.2)
      a[2 * r]     = As[mrow * 256 + ka];
      a[2 * r + 1] = As[mrow * 256 + ka + 1];
      int kq = k0 + 2 * r + ((r >= 4) ? 8 : 0) + hiB;   // 16-bit B layout, same k order as A
      b[2 * r]     = Wp[((size_t)mode * 256 + kq) * COUT + o0 + ncol];
      b[2 * r + 1] = Wp[((size_t)mode * 256 + kq + 1) * COUT + o0 + ncol];
    }
    acc = wmma_bf16(a, b, acc);
  }
  for (int r = 0; r < 8; ++r) {
    int row = r + hiA;
    int o = o0 + ncol, bb = row & 7;
    size_t addr = (((size_t)(bb * COUT + o)) * M1 + mm) * NP + nn;
    if (row < 8) Ysr[addr] = acc[r];
    else         Ysi[addr] = acc[r];
  }
}

// ------------------------------------------------- stage 4: H-direction inverse DFT (complex)
__global__ void __launch_bounds__(256) k_stage4(const float* __restrict__ Ysr,
                                                const float* __restrict__ Ysi,
                                                const float* __restrict__ Ghr,
                                                const float* __restrict__ Ghi,
                                                float* __restrict__ Yhr,
                                                float* __restrict__ Yhi) {
  int bo = blockIdx.x;
  const float* brb = Ysr + (size_t)bo * M1 * NP;
  const float* bib = Ysi + (size_t)bo * M1 * NP;
  int t = threadIdx.x, lane = t & 31, wave = t >> 5;
  int m = lane & 15, hi = lane >> 4, kb = hi * 2, n = lane & 15;
  for (int mt = wave * 2; mt < wave * 2 + 2; ++mt) {
    int h0 = mt * 16;
    for (int nt = 0; nt < 3; ++nt) {
      v8f arr = {}, aii = {}, ari = {}, air = {};
      for (int k0 = 0; k0 < M1; k0 += 4) {
        v2f ar, ai, br, bi2;
        ar.x = Ghr[(h0 + m) * M1 + k0 + kb]; ar.y = Ghr[(h0 + m) * M1 + k0 + kb + 1];
        ai.x = Ghi[(h0 + m) * M1 + k0 + kb]; ai.y = Ghi[(h0 + m) * M1 + k0 + kb + 1];
        br.x = brb[(k0 + kb) * NP + nt * 16 + n];  br.y = brb[(k0 + kb + 1) * NP + nt * 16 + n];
        bi2.x = bib[(k0 + kb) * NP + nt * 16 + n]; bi2.y = bib[(k0 + kb + 1) * NP + nt * 16 + n];
        arr = wmma_f32(ar, br, arr);  aii = wmma_f32(ai, bi2, aii);
        ari = wmma_f32(ar, bi2, ari); air = wmma_f32(ai, br, air);
      }
      for (int r = 0; r < 8; ++r) {
        int hc = h0 + r + hi * 8, nc = nt * 16 + n;
        size_t addr = ((size_t)bo * HH + hc) * NP + nc;
        Yhr[addr] = arr[r] - aii[r];
        Yhi[addr] = ari[r] + air[r];
      }
    }
  }
}

// --------------- stage 5: fused  skip_w@x_slice + Yh_slice@G2 + bias -> GELU -> out
// One workgroup per (b,h); 8 waves own the 8 o-tiles; 16 w-tiles each.
__global__ void __launch_bounds__(256) k_stage5(const float* __restrict__ x,
                                                const float* __restrict__ skw,
                                                const float* __restrict__ Yhr,
                                                const float* __restrict__ Yhi,
                                                const float* __restrict__ G2,
                                                const float* __restrict__ bias,
                                                float* __restrict__ out) {
  int b = blockIdx.x >> 8, h = blockIdx.x & 255;
  int t = threadIdx.x, lane = t & 31, wave = t >> 5;
  int m = lane & 15, hi = lane >> 4, kb = hi * 2, n = lane & 15;
  int m0 = wave * 16;
  const float* xb = x + ((size_t)b * CIN) * HH * WW + (size_t)h * WW;  // xb[k*H*W + w]
  size_t yrow = (((size_t)(b * COUT + m0 + m)) * HH + h) * NP;         // Yh row for this lane
  for (int nt = 0; nt < 16; ++nt) {
    int w0 = nt * 16;
    // pull next w-tile of x toward the WGP while this tile's WMMAs run
    if (nt < 15)
      __builtin_prefetch(&xb[(size_t)kb * HH * WW + w0 + 16 + n], 0, 1);
    v8f acc = {};
    // skip connection: skip_w (128x128) @ x[b,:,h,w0..w0+15]
    for (int k0 = 0; k0 < CIN; k0 += 4) {
      v2f a, bb;
      a.x = skw[(m0 + m) * CIN + k0 + kb];
      a.y = skw[(m0 + m) * CIN + k0 + kb + 1];
      bb.x = xb[(size_t)(k0 + kb) * HH * WW + w0 + n];
      bb.y = xb[(size_t)(k0 + kb + 1) * HH * WW + w0 + n];
      acc = wmma_f32(a, bb, acc);
    }
    // inverse rDFT along W: [Yr|Yi] (128x96) @ G2 (96x256)
    for (int k0 = 0; k0 < 96; k0 += 4) {
      int j = k0 + kb;   // even, so j and j+1 never straddle the re/im split
      v2f a, bb;
      a.x = (j < NP)     ? Yhr[yrow + j]     : Yhi[yrow + j - NP];
      a.y = (j + 1 < NP) ? Yhr[yrow + j + 1] : Yhi[yrow + j + 1 - NP];
      bb.x = G2[j * 256 + w0 + n];
      bb.y = G2[(j + 1) * 256 + w0 + n];
      acc = wmma_f32(a, bb, acc);
    }
    for (int r = 0; r < 8; ++r) {
      int oc = m0 + r + hi * 8, wc = w0 + n;
      float v = gelu_exact(acc[r] + bias[oc]);
      out[(((size_t)(b * COUT + oc)) * HH + h) * WW + wc] = v;
    }
  }
}

// ----------------------------------------------------------------------------- launcher
extern "C" void kernel_launch(void* const* d_in, const int* in_sizes, int n_in,
                              void* d_out, int out_size, void* d_ws, size_t ws_size,
                              hipStream_t stream) {
  (void)in_sizes; (void)n_in; (void)out_size; (void)ws_size;
  const float* x    = (const float*)d_in[0];
  const float* wr   = (const float*)d_in[1];
  const float* wi   = (const float*)d_in[2];
  const float* bias = (const float*)d_in[3];
  const float* skw  = (const float*)d_in[4];
  float* out = (float*)d_out;
  char*  ws  = (char*)d_ws;

  size_t off = 0;
  float* F1  = (float*)(ws + off); off += (size_t)256 * 96 * 4;
  float* Fhr = (float*)(ws + off); off += (size_t)M1 * 256 * 4;
  float* Fhi = (float*)(ws + off); off += (size_t)M1 * 256 * 4;
  float* Ghr = (float*)(ws + off); off += (size_t)256 * M1 * 4;
  float* Ghi = (float*)(ws + off); off += (size_t)256 * M1 * 4;
  float* G2  = (float*)(ws + off); off += (size_t)96 * 256 * 4;
  off = (off + 255) & ~(size_t)255;
  __bf16* Wp = (__bf16*)(ws + off); off += (size_t)NMODE * 256 * COUT * 2;   // 132 MiB
  const size_t XW = (size_t)BATCH * CIN * HH * NP;   // 12.58M floats
  const size_t XS = (size_t)BATCH * CIN * M1 * NP;   //  3.15M floats
  float* Xwr = (float*)(ws + off); off += XW * 4;
  float* Xwi = (float*)(ws + off); off += XW * 4;
  float* Xsr = (float*)(ws + off); off += XS * 4;
  float* Xsi = (float*)(ws + off); off += XS * 4;
  float* Ysr = (float*)(ws + off); off += XS * 4;
  float* Ysi = (float*)(ws + off); off += XS * 4;
  float* Yhr = Xwr;   // Xw is dead after stage 2 -> reuse for Yh
  float* Yhi = Xwi;

  k_tables<<<1, 256, 0, stream>>>(F1, Fhr, Fhi, Ghr, Ghi, G2);
  k_pack<<<dim3(256, 33), 256, 0, stream>>>(wr, wi, Wp);
  k_stage1<<<(BATCH * CIN * HH) / 32, 64, 0, stream>>>(x, F1, Xwr, Xwi);
  k_stage2<<<BATCH * CIN, 128, 0, stream>>>(Xwr, Xwi, Fhr, Fhi, Xsr, Xsi);
  hipMemsetAsync(Ysr, 0, XS * 4, stream);   // cols 33..47 must be zero for stage 4
  hipMemsetAsync(Ysi, 0, XS * 4, stream);
  k_stage3<<<NMODE, 256, 0, stream>>>(Xsr, Xsi, Wp, Ysr, Ysi);
  k_stage4<<<BATCH * COUT, 256, 0, stream>>>(Ysr, Ysi, Ghr, Ghi, Yhr, Yhi);
  k_stage5<<<BATCH * HH, 256, 0, stream>>>(x, skw, Yhr, Yhi, G2, bias, out);
}
